// BlockCausalTransformer_87625922773471
// MI455X (gfx1250) — compile-verified, matrix operand order checked by reading, not run on hardware
//
#include <hip/hip_runtime.h>

// ---------------------------------------------------------------------------
// MI455X (gfx1250) block-causal transformer forward.
// B=4, S=1024, D=1024, H=16, DH=64, DFF=4096, L=4, BLK=64, M = B*S = 4096.
// Strategy: bf16 WMMA (v_wmma_f32_16x16x32_bf16) for all GEMMs + flash attn.
// ---------------------------------------------------------------------------

typedef __attribute__((ext_vector_type(16))) __bf16 v16bf;
typedef __attribute__((ext_vector_type(8)))  float  v8f;
typedef __attribute__((ext_vector_type(4)))  int    v4i;

#define M_TOK   4096
#define D_MODEL 1024
#define DFF_    4096
#define NH      16
#define DH_     64
#define S_LEN   1024
#define NB      4

// --------------------------- helpers ---------------------------------------

__device__ __forceinline__ unsigned pk_bf16(float a, float b) {
  // round-to-nearest-even f32 -> bf16, packed pair (a in low 16, b in high 16)
  unsigned ua = __float_as_uint(a), ub = __float_as_uint(b);
  ua += 0x7fffu + ((ua >> 16) & 1u);
  ub += 0x7fffu + ((ub >> 16) & 1u);
  return (ua >> 16) | (ub & 0xffff0000u);
}

__device__ __forceinline__ v8f wmma_bf16(v16bf a, v16bf b, v8f c) {
  return __builtin_amdgcn_wmma_f32_16x16x32_bf16(
      /*neg_a=*/false, a, /*neg_b=*/false, b,
      /*c_mod=*/(short)0, c, /*reuse_a=*/false, /*reuse_b=*/false);
}

// 16-bit A-fragment loader (also used for B since we keep both operands
// row-major [row][K] in LDS; B's "row" is the output column n).
// ISA layout: lane L -> row = L%16, half = L/16; VGPR v in 0..3 holds
// K = 8*half + 2v,+1 ; v in 4..7 holds K = 8*half + 16 + 2(v-4),+1.
// -> two contiguous 16B loads at byte offsets (row*ld + k0 + 8*half)*2 and +32.
__device__ __forceinline__ v16bf ld_frag(const __bf16* base, int row0, int k0, int ld) {
  int lane = threadIdx.x & 31;
  const uint4* p = (const uint4*)(base + (size_t)(row0 + (lane & 15)) * ld
                                       + k0 + ((lane >> 4) << 3));
  union { uint4 q[2]; v16bf v; } u;
  u.q[0] = p[0];
  u.q[1] = p[2];   // +32 bytes == +16 bf16 elements (K+16 group)
  return u.v;
}

// async global->LDS 16B copy (CDNA5 GLOBAL_LOAD_ASYNC_TO_LDS_B128) with a
// portable fallback; guarded so it compiles on both toolchains.
// Probe-confirmed param types: (v4i __device__*, v4i AS(3)*, Ii offset, Ii cpol).
#if __has_builtin(__builtin_amdgcn_global_load_async_to_lds_b128)
#define HAVE_ASYNC_LDS 1
typedef __attribute__((address_space(1))) v4i gas_v4i;
typedef __attribute__((address_space(3))) v4i las_v4i;
#endif

__device__ __forceinline__ void cp16_g2l(void* lds, const void* g) {
#ifdef HAVE_ASYNC_LDS
  __builtin_amdgcn_global_load_async_to_lds_b128(
      (gas_v4i*)(uintptr_t)g,
      (las_v4i*)(unsigned)(uintptr_t)lds,
      0, 0);
#else
  *(uint4*)lds = *(const uint4*)g;
#endif
}

__device__ __forceinline__ void async_wait() {
#ifdef HAVE_ASYNC_LDS
#if __has_builtin(__builtin_amdgcn_s_wait_asynccnt)
  __builtin_amdgcn_s_wait_asynccnt(0);
#else
  asm volatile("s_wait_asynccnt 0" ::: "memory");
#endif
#endif
}

// --------------------------- convert kernels --------------------------------

__global__ __launch_bounds__(256) void k_convert_bf16(
    const float* __restrict__ in, __bf16* __restrict__ out, int n) {
  int i = (blockIdx.x * 256 + threadIdx.x) * 4;
  if (i < n) {
    float4 v = *(const float4*)(in + i);
    *(uint2*)(out + i) = make_uint2(pk_bf16(v.x, v.y), pk_bf16(v.z, v.w));
  }
}

__global__ __launch_bounds__(256) void k_prep_x(
    const float* __restrict__ in, float* __restrict__ xf,
    __bf16* __restrict__ xb, int n) {
  int i = (blockIdx.x * 256 + threadIdx.x) * 4;
  if (i < n) {
    float4 v = *(const float4*)(in + i);
    *(float4*)(xf + i) = v;
    *(uint2*)(xb + i) = make_uint2(pk_bf16(v.x, v.y), pk_bf16(v.z, v.w));
  }
}

// --------------------------- GEMM: D = A @ W^T + bias -----------------------
// A: [M,K] bf16 row-major.  W: [N,K] bf16 row-major (so W^T cols == W rows).
// 256 threads = 8 waves; block tile 128(M) x 64(N); K step 64.
// Wave (wm,wn) computes 32x32 = 2x2 WMMA tiles.
// flags: bit0 = relu, bit1 = write bf16 (else f32).

#define BM 128
#define BN 64
#define BK 64

__global__ __launch_bounds__(256) void k_gemm_wmma(
    const __bf16* __restrict__ A, const __bf16* __restrict__ W,
    const float* __restrict__ bias,
    float* __restrict__ Cf, __bf16* __restrict__ Cb,
    int M, int N, int K, int flags) {
  __shared__ __align__(16) __bf16 As[BM * BK];
  __shared__ __align__(16) __bf16 Bs[BN * BK];

  const int tid  = threadIdx.x;
  const int wave = tid >> 5;
  const int lane = tid & 31;
  const int bm   = blockIdx.y * BM;
  const int bn   = blockIdx.x * BN;
  const int wm   = (wave & 3) * 32;
  const int wn   = (wave >> 2) * 32;

  v8f acc[2][2] = {};

  for (int k0 = 0; k0 < K; k0 += BK) {
    // stage A tile: 128 rows x 128 B/row = 1024 x 16B chunks
    const char* Ag = (const char*)A;
    const char* Wg = (const char*)W;
#pragma unroll
    for (int it = 0; it < 4; ++it) {
      int c = tid + it * 256;
      int row = c >> 3, colb = (c & 7) * 16;
      cp16_g2l((char*)As + c * 16,
               Ag + ((size_t)(bm + row) * K + k0) * 2 + colb);
    }
    // stage W tile: 64 rows -> 512 chunks
#pragma unroll
    for (int it = 0; it < 2; ++it) {
      int c = tid + it * 256;
      int row = c >> 3, colb = (c & 7) * 16;
      cp16_g2l((char*)Bs + c * 16,
               Wg + ((size_t)(bn + row) * K + k0) * 2 + colb);
    }
    async_wait();
    __syncthreads();

#pragma unroll
    for (int kk = 0; kk < BK; kk += 32) {
      v16bf a0 = ld_frag(As, wm,      kk, BK);
      v16bf a1 = ld_frag(As, wm + 16, kk, BK);
      v16bf b0 = ld_frag(Bs, wn,      kk, BK);
      v16bf b1 = ld_frag(Bs, wn + 16, kk, BK);
      acc[0][0] = wmma_bf16(a0, b0, acc[0][0]);
      acc[0][1] = wmma_bf16(a0, b1, acc[0][1]);
      acc[1][0] = wmma_bf16(a1, b0, acc[1][0]);
      acc[1][1] = wmma_bf16(a1, b1, acc[1][1]);
    }
    __syncthreads();
  }

  // epilogue: C/D layout -> lane: n = lane%16, rows m = r + 8*(lane/16)
  const int nl   = lane & 15;
  const int half = lane >> 4;
#pragma unroll
  for (int j = 0; j < 2; ++j) {
    int gn = bn + wn + 16 * j + nl;
    float bv = bias[gn];
#pragma unroll
    for (int i = 0; i < 2; ++i) {
#pragma unroll
      for (int r = 0; r < 8; ++r) {
        int gm = bm + wm + 16 * i + r + 8 * half;
        float v = acc[i][j][r] + bv;
        if (flags & 1) v = fmaxf(v, 0.0f);
        size_t idx = (size_t)gm * N + gn;
        if (flags & 2) Cb[idx] = (__bf16)v;
        else           Cf[idx] = v;
      }
    }
  }
}

// --------------------------- flash attention --------------------------------
// One workgroup (128 threads = 4 waves) per (b, h, 64-query block).
// Block-causal mask with BLK=64 == our q-tile size -> k-blocks 0..qb are
// fully visible, nothing else: loop kb in [0, qb].
// Wave w owns query rows [16w,16w+16); O accum = 4 tiles of 16x16 (dh = 64).

__global__ __launch_bounds__(128) void k_attn(
    const float* __restrict__ qkv,   // [B,S,3D] f32
    __bf16* __restrict__ o) {        // [B,S,D] bf16 (per-head concat)
  __shared__ __align__(16) __bf16 Qs[64 * 64];
  __shared__ __align__(16) __bf16 Ks[64 * 64];
  __shared__ __align__(16) __bf16 Vt[64 * 64];   // transposed: [dh][key]
  __shared__ __align__(16) __bf16 Ps[64 * 64];
  __shared__ float Ss[64 * 64];
  __shared__ float mrow[64], lrow[64], srow[64];

  const int qb = blockIdx.x & 15;
  const int h  = (blockIdx.x >> 4) & 15;
  const int b  = blockIdx.x >> 8;
  const float scale = 0.125f;               // 1/sqrt(DH)
  const int tid = threadIdx.x, wave = tid >> 5, lane = tid & 31;
  const size_t rs = 3 * D_MODEL;            // qkv row stride

  const float* qbase = qkv + ((size_t)b * S_LEN + qb * 64) * rs + h * DH_;

  // stage Q (f32 -> bf16): 64x64 = 1024 float4
  for (int c = tid; c < 1024; c += 128) {
    int r = c >> 4, col = (c & 15) * 4;
    float4 v = *(const float4*)(qbase + (size_t)r * rs + col);
    *(uint2*)(&Qs[r * 64 + col]) = make_uint2(pk_bf16(v.x, v.y), pk_bf16(v.z, v.w));
  }
  if (tid < 64) { mrow[tid] = -3.0e38f; lrow[tid] = 0.0f; }
  v8f oacc[4] = {};
  __syncthreads();

  for (int kb = 0; kb <= qb; ++kb) {
    const float* kbase = qkv + ((size_t)b * S_LEN + kb * 64) * rs + D_MODEL + h * DH_;
    const float* vbase = kbase + D_MODEL;
    for (int c = tid; c < 1024; c += 128) {
      int r = c >> 4, col = (c & 15) * 4;
      float4 v = *(const float4*)(kbase + (size_t)r * rs + col);
      *(uint2*)(&Ks[r * 64 + col]) = make_uint2(pk_bf16(v.x, v.y), pk_bf16(v.z, v.w));
    }
    // V transposed into LDS: Vt[dh][key]; pack key pairs
    for (int c = tid; c < 2048; c += 128) {
      int col = c >> 5, kp = (c & 31) * 2;
      float v0 = vbase[(size_t)kp * rs + col];
      float v1 = vbase[(size_t)(kp + 1) * rs + col];
      *(unsigned*)(&Vt[col * 64 + kp]) = pk_bf16(v0, v1);
    }
    __syncthreads();

    // S = Q @ K^T  (16 query rows per wave x 64 keys)
#pragma unroll
    for (int t = 0; t < 4; ++t) {
      v8f s = {};
      s = wmma_bf16(ld_frag(Qs, 16 * wave, 0, 64),  ld_frag(Ks, 16 * t, 0, 64),  s);
      s = wmma_bf16(ld_frag(Qs, 16 * wave, 32, 64), ld_frag(Ks, 16 * t, 32, 64), s);
      int n = 16 * t + (lane & 15), half = lane >> 4;
#pragma unroll
      for (int r = 0; r < 8; ++r)
        Ss[(16 * wave + r + 8 * half) * 64 + n] = s[r];
    }
    __syncthreads();

    // online softmax: one thread per query row
    if (tid < 64) {
      float mo = mrow[tid], rm = -3.0e38f;
      for (int c = 0; c < 64; ++c) rm = fmaxf(rm, Ss[tid * 64 + c] * scale);
      float mn = fmaxf(mo, rm);
      float sc = __expf(mo - mn);
      float ls = 0.0f;
      for (int c = 0; c < 64; ++c) {
        float p = __expf(Ss[tid * 64 + c] * scale - mn);
        ls += p;
        Ps[tid * 64 + c] = (__bf16)p;
      }
      lrow[tid] = lrow[tid] * sc + ls;
      mrow[tid] = mn;
      srow[tid] = sc;
    }
    __syncthreads();

    // rescale running O, then O += P @ V
    {
      int half = lane >> 4;
#pragma unroll
      for (int r = 0; r < 8; ++r) {
        float sc = srow[16 * wave + r + 8 * half];
#pragma unroll
        for (int t = 0; t < 4; ++t) oacc[t][r] *= sc;
      }
    }
#pragma unroll
    for (int t = 0; t < 4; ++t) {
      oacc[t] = wmma_bf16(ld_frag(Ps, 16 * wave, 0, 64),  ld_frag(Vt, 16 * t, 0, 64),  oacc[t]);
      oacc[t] = wmma_bf16(ld_frag(Ps, 16 * wave, 32, 64), ld_frag(Vt, 16 * t, 32, 64), oacc[t]);
    }
    __syncthreads();
  }

  // finalize: O /= l, write bf16
  {
    int half = lane >> 4, nl = lane & 15;
#pragma unroll
    for (int t = 0; t < 4; ++t)
#pragma unroll
      for (int r = 0; r < 8; ++r) {
        int m = 16 * wave + r + 8 * half;
        int row = qb * 64 + m;
        float v = oacc[t][r] / lrow[m];
        o[((size_t)b * S_LEN + row) * D_MODEL + h * DH_ + 16 * t + nl] = (__bf16)v;
      }
  }
}

// --------------------------- residual + LayerNorm ---------------------------
// One block (256 thr) per row of 1024; writes f32 (next residual stream) and
// bf16 (next GEMM's A operand).

__global__ __launch_bounds__(256) void k_add_ln(
    const float* __restrict__ x, const float* __restrict__ d,
    const float* __restrict__ w, const float* __restrict__ bb,
    float* __restrict__ xout, __bf16* __restrict__ xbf) {
  __shared__ float red[256];
  const int row = blockIdx.x, tid = threadIdx.x;
  const float* xr = x + (size_t)row * D_MODEL;
  const float* dr = d + (size_t)row * D_MODEL;

  float v[4];
  float s = 0.0f;
#pragma unroll
  for (int i = 0; i < 4; ++i) {
    int c = tid + 256 * i;
    v[i] = xr[c] + dr[c];
    s += v[i];
  }
  red[tid] = s; __syncthreads();
  for (int off = 128; off > 0; off >>= 1) {
    if (tid < off) red[tid] += red[tid + off];
    __syncthreads();
  }
  float mu = red[0] * (1.0f / D_MODEL);
  __syncthreads();

  float s2 = 0.0f;
#pragma unroll
  for (int i = 0; i < 4; ++i) { float t = v[i] - mu; s2 += t * t; }
  red[tid] = s2; __syncthreads();
  for (int off = 128; off > 0; off >>= 1) {
    if (tid < off) red[tid] += red[tid + off];
    __syncthreads();
  }
  float rstd = rsqrtf(red[0] * (1.0f / D_MODEL) + 1e-5f);

#pragma unroll
  for (int i = 0; i < 4; ++i) {
    int c = tid + 256 * i;
    float y = (v[i] - mu) * rstd * w[c] + bb[c];
    xout[(size_t)row * D_MODEL + c] = y;
    xbf [(size_t)row * D_MODEL + c] = (__bf16)y;
  }
}

// --------------------------- host orchestration -----------------------------

extern "C" void kernel_launch(void* const* d_in, const int* in_sizes, int n_in,
                              void* d_out, int out_size, void* d_ws, size_t ws_size,
                              hipStream_t stream) {
  const float* x_in = (const float*)d_in[0];
  const float* inw  = (const float*)d_in[1];
  const float* inb  = (const float*)d_in[2];
  const float* outw = (const float*)d_in[3];
  const float* outb = (const float*)d_in[4];
  const float* ln1w = (const float*)d_in[5];
  const float* ln1b = (const float*)d_in[6];
  const float* l1w  = (const float*)d_in[7];
  const float* l1b  = (const float*)d_in[8];
  const float* l2w  = (const float*)d_in[9];
  const float* l2b  = (const float*)d_in[10];
  const float* ln2w = (const float*)d_in[11];
  const float* ln2b = (const float*)d_in[12];

  // workspace layout (128 MB total)
  char* ws = (char*)d_ws;
  size_t off = 0;
  float*  xc    = (float*) (ws + off); off += (size_t)M_TOK * D_MODEL * 4;   // 16 MB
  float*  qkvb  = (float*) (ws + off); off += (size_t)M_TOK * 3 * D_MODEL * 4; // 48 MB
  float*  projb = (float*) (ws + off); off += (size_t)M_TOK * D_MODEL * 4;   // 16 MB
  __bf16* xa    = (__bf16*)(ws + off); off += (size_t)M_TOK * D_MODEL * 2;   // 8 MB
  __bf16* hbuf  = (__bf16*)(ws + off); off += (size_t)M_TOK * DFF_ * 2;      // 32 MB
  __bf16* wbuf  = (__bf16*)(ws + off); off += (size_t)DFF_ * D_MODEL * 2;    // 8 MB
  (void)ws_size; (void)in_sizes; (void)n_in; (void)out_size;

  // x -> x_f32 copy + bf16 copy
  k_prep_x<<<4096, 256, 0, stream>>>(x_in, xc, xa, M_TOK * D_MODEL);

  for (int l = 0; l < 4; ++l) {
    // ---- QKV projection: [4096,1024] @ [3072,1024]^T ----
    k_convert_bf16<<<3072, 256, 0, stream>>>(inw + (size_t)l * 3 * D_MODEL * D_MODEL,
                                             wbuf, 3 * D_MODEL * D_MODEL);
    k_gemm_wmma<<<dim3(3 * D_MODEL / BN, M_TOK / BM), 256, 0, stream>>>(
        xa, wbuf, inb + (size_t)l * 3 * D_MODEL, qkvb, nullptr,
        M_TOK, 3 * D_MODEL, D_MODEL, 0);

    // ---- block-causal flash attention (writes bf16 heads into xa) ----
    k_attn<<<NB * NH * (S_LEN / 64), 128, 0, stream>>>(qkvb, xa);

    // ---- output projection ----
    k_convert_bf16<<<1024, 256, 0, stream>>>(outw + (size_t)l * D_MODEL * D_MODEL,
                                             wbuf, D_MODEL * D_MODEL);
    k_gemm_wmma<<<dim3(D_MODEL / BN, M_TOK / BM), 256, 0, stream>>>(
        xa, wbuf, outb + (size_t)l * D_MODEL, projb, nullptr,
        M_TOK, D_MODEL, D_MODEL, 0);

    // ---- residual + LN1 ----
    k_add_ln<<<M_TOK, 256, 0, stream>>>(xc, projb,
        ln1w + (size_t)l * D_MODEL, ln1b + (size_t)l * D_MODEL, xc, xa);

    // ---- FFN up (relu, bf16 out) ----
    k_convert_bf16<<<4096, 256, 0, stream>>>(l1w + (size_t)l * DFF_ * D_MODEL,
                                             wbuf, DFF_ * D_MODEL);
    k_gemm_wmma<<<dim3(DFF_ / BN, M_TOK / BM), 256, 0, stream>>>(
        xa, wbuf, l1b + (size_t)l * DFF_, nullptr, hbuf,
        M_TOK, DFF_, D_MODEL, /*relu|bf16*/ 3);

    // ---- FFN down ----
    k_convert_bf16<<<4096, 256, 0, stream>>>(l2w + (size_t)l * D_MODEL * DFF_,
                                             wbuf, D_MODEL * DFF_);
    k_gemm_wmma<<<dim3(D_MODEL / BN, M_TOK / BM), 256, 0, stream>>>(
        hbuf, wbuf, l2b + (size_t)l * D_MODEL, projb, nullptr,
        M_TOK, D_MODEL, DFF_, 0);

    // ---- residual + LN2 (last layer writes d_out) ----
    float* xo = (l == 3) ? (float*)d_out : xc;
    k_add_ln<<<M_TOK, 256, 0, stream>>>(xc, projb,
        ln2w + (size_t)l * D_MODEL, ln2b + (size_t)l * D_MODEL, xo, xa);
  }
}
